// SRGNN_46351287058914
// MI455X (gfx1250) — matrix operand order repaired and checked by reading, not compiled, not run on hardware
//
#include <hip/hip_runtime.h>
#include <cstddef>
#include <cstdint>

// ---------------------------------------------------------------------------
// SR-GNN forward on MI455X (gfx1250, wave32).
// Heavy GEMMs use v_wmma_f32_16x16x32_bf16. Final logits GEMM is HBM-write
// bound (~205MB @ 23.3TB/s ~= 9us); bf16 embedding copy (25.6MB) is L2
// resident (192MB L2), so WMMA keeps compute far under the write-bound time.
// ---------------------------------------------------------------------------

typedef __bf16 bf16;
typedef __attribute__((ext_vector_type(8)))  bf16  bf16x8;
typedef __attribute__((ext_vector_type(16))) bf16  v16bf;
typedef __attribute__((ext_vector_type(8)))  float v8f;

#define Bsz   512
#define Lsz   50
#define Dsz   128
#define Vsz   100000
#define Vp1   100001
#define BL    (Bsz * Lsz)          // 25600

// ---------------- workspace layout (bytes, 256-aligned) --------------------
#define OFF_EMB     ((size_t)0)                // (V+1)*128 bf16 = 25,600,256
#define OFF_WPACK   ((size_t)25600512)         // 10 * 128*128 bf16
#define OFF_AIN     ((size_t)25928192)         // B*50*50 f32
#define OFF_AOUT    ((size_t)31048192)
#define OFF_H       ((size_t)36168192)         // BL*128 f32
#define OFF_HBF     ((size_t)49275392)         // BL*128 bf16
#define OFF_HIO     ((size_t)55828992)         // BL*256 f32  [Hin|Hout]
#define OFF_M       ((size_t)82043392)         // BL*128 f32
#define OFF_MBF     ((size_t)95150592)         // BL*128 bf16
#define OFF_GM      ((size_t)101704192)        // BL*384 f32  [mWz|mWr|mWh]
#define OFF_GH      ((size_t)141025792)        // BL*256 f32  [hUz|hUr]
#define OFF_RHBF    ((size_t)167240192)        // BL*128 bf16
#define OFF_G6      ((size_t)173793792)        // BL*128 f32  (r*h)@U_h
// aliases (GM/GH dead after gates2):
#define OFF_SEQH    OFF_GM                     // BL*128 f32
#define OFF_SEQHBF  (OFF_GM + (size_t)13107200)// BL*128 bf16
#define OFF_Q       OFF_GH                     // BL*128 f32
#define OFF_LASTH   ((size_t)186900992)        // B*128 f32
#define OFF_LASTHBF ((size_t)187163136)        // B*128 bf16
#define OFF_K1      ((size_t)187294208)        // B*128 f32
#define OFF_SG      ((size_t)187556352)        // B*128 f32
#define OFF_SREPBF  ((size_t)187818496)        // B*128 bf16
#define OFF_INV     ((size_t)187949568)        // BL int
#define OFF_LEN     ((size_t)188051968)        // B int
#define OFF_LASTN   ((size_t)188054016)        // B int
// total ~188 MB of d_ws

// ---------------- WMMA fragment helpers (contiguous k per lane half) ---------------
__device__ __forceinline__ v16bf cat8(bf16x8 lo, bf16x8 hi) {
  return __builtin_shufflevector(lo, hi, 0,1,2,3,4,5,6,7,8,9,10,11,12,13,14,15);
}

// A: 16x32 bf16, row-major, lda elems. lane<16: K {0..15}; lane>=16: {16..31}
__device__ __forceinline__ v16bf load_a_frag(const bf16* __restrict__ A, int lda,
                                             int m0, int k0, int lane) {
  const bf16* p = A + (size_t)(m0 + (lane & 15)) * lda + k0 + ((lane >> 4) << 4);
  bf16x8 lo = *(const bf16x8*)p;
  bf16x8 hi = *(const bf16x8*)(p + 8);
  return cat8(lo, hi);
}

// B: 32x16 bf16 from Bt (N rows x K cols, row-major). lane<16: K k0..k0+15 of col n;
// lane>=16: K k0+16..k0+31.
__device__ __forceinline__ v16bf load_b_frag(const bf16* __restrict__ Bt, int ldb,
                                             int n0, int k0, int lane) {
  const bf16* p = Bt + (size_t)(n0 + (lane & 15)) * ldb + k0 + ((lane >> 4) << 4);
  bf16x8 lo = *(const bf16x8*)p;
  bf16x8 hi = *(const bf16x8*)(p + 8);
  return cat8(lo, hi);
}

#define WMMA_BF16(a, b, c) \
  __builtin_amdgcn_wmma_f32_16x16x32_bf16(false, (a), false, (b), (short)0, (c), false, false)

// ---------------- generic GEMM: C[M,N] = A[M,128] @ Bt[N,128]^T ------------
// one wave per (m-tile, 4 n-tile strip); A (K=128) held in registers.
__global__ void wmma_gemm_kernel(const bf16* __restrict__ A, const bf16* __restrict__ Bt,
                                 float* __restrict__ C, int M, int N) {
  const int lane = threadIdx.x;            // blockDim.x == 32
  const int m0 = blockIdx.x * 16;
  const int ntiles = N >> 4;
  v16bf a0 = load_a_frag(A, 128, m0,  0, lane);
  v16bf a1 = load_a_frag(A, 128, m0, 32, lane);
  v16bf a2 = load_a_frag(A, 128, m0, 64, lane);
  v16bf a3 = load_a_frag(A, 128, m0, 96, lane);
  for (int t = 0; t < 4; ++t) {
    int nt = blockIdx.y * 4 + t;
    if (nt >= ntiles) break;               // uniform across wave
    int n0 = nt * 16;
    v8f c = {};
    c = WMMA_BF16(a0, load_b_frag(Bt, 128, n0,  0, lane), c);
    c = WMMA_BF16(a1, load_b_frag(Bt, 128, n0, 32, lane), c);
    c = WMMA_BF16(a2, load_b_frag(Bt, 128, n0, 64, lane), c);
    c = WMMA_BF16(a3, load_b_frag(Bt, 128, n0, 96, lane), c);
    float* cp = C + (size_t)(m0 + ((lane >> 4) << 3)) * N + n0 + (lane & 15);
#pragma unroll
    for (int r = 0; r < 8; ++r) cp[(size_t)r * N] = c[r];
  }
}

// ---------------- logits: out[512, 100001] = srep @ emb^T ------------------
// wave per vocab n-tile; B (one emb column strip) in registers, loops 32 m-tiles.
__global__ void logits_kernel(const bf16* __restrict__ srep, const bf16* __restrict__ emb,
                              float* __restrict__ out) {
  const int lane = threadIdx.x;
  const int n0 = blockIdx.x * 16;
  const int col = n0 + (lane & 15);
  const int colc = col > Vsz ? Vsz : col;  // clamp loads on ragged last tile
  v16bf b0 = load_b_frag(emb, 128, 0,  0, (lane & 16)) ;
  // (note: load_b_frag uses n0+(lane&15); redo with clamped column)
  {
    const bf16* p = emb + (size_t)colc * 128 + ((lane >> 4) << 4);
    b0 = cat8(*(const bf16x8*)(p +  0), *(const bf16x8*)(p +  8));
  }
  const bf16* p = emb + (size_t)colc * 128 + ((lane >> 4) << 4);
  v16bf b1 = cat8(*(const bf16x8*)(p + 32), *(const bf16x8*)(p + 40));
  v16bf b2 = cat8(*(const bf16x8*)(p + 64), *(const bf16x8*)(p + 72));
  v16bf b3 = cat8(*(const bf16x8*)(p + 96), *(const bf16x8*)(p + 104));
  for (int mt = 0; mt < Bsz / 16; ++mt) {
    int m0 = mt * 16;
    v8f c = {};
    c = WMMA_BF16(load_a_frag(srep, 128, m0,  0, lane), b0, c);
    c = WMMA_BF16(load_a_frag(srep, 128, m0, 32, lane), b1, c);
    c = WMMA_BF16(load_a_frag(srep, 128, m0, 64, lane), b2, c);
    c = WMMA_BF16(load_a_frag(srep, 128, m0, 96, lane), b3, c);
    if (col <= Vsz) {
      float* cp = out + (size_t)(m0 + ((lane >> 4) << 3)) * Vp1 + col;
#pragma unroll
      for (int r = 0; r < 8; ++r) cp[(size_t)r * Vp1] = c[r];
    }
  }
}

// ---------------- small kernels --------------------------------------------
__global__ void packT_kernel(const float* __restrict__ W, bf16* __restrict__ out) {
  int n = blockIdx.x, k = threadIdx.x;     // Bt[n][k] = W[k][n]
  out[(size_t)n * 128 + k] = (bf16)W[(size_t)k * 128 + n];
}

__global__ void cvt_bf16_kernel(const float* __restrict__ in, bf16* __restrict__ out, size_t n) {
  size_t i = (size_t)blockIdx.x * blockDim.x + threadIdx.x;
  if (i < n) out[i] = (bf16)in[i];
}

__global__ void preprocess_kernel(const int* __restrict__ seq, const float* __restrict__ emb,
                                  float* __restrict__ Ain, float* __restrict__ Aout,
                                  float* __restrict__ h, bf16* __restrict__ hbf,
                                  int* __restrict__ inv_g, int* __restrict__ len_g,
                                  int* __restrict__ lastn_g) {
  const int b = blockIdx.x, tid = threadIdx.x;   // 64 threads
  __shared__ int s_seq[Lsz], s_inv[Lsz], s_comp[Lsz];
  __shared__ unsigned char s_val[Lsz];
  __shared__ float sAin[Lsz][Lsz], sAout[Lsz][Lsz];
  __shared__ float rin[Lsz], rout[Lsz];
  for (int l = tid; l < Lsz; l += 64) s_seq[l] = seq[b * Lsz + l];
  __syncthreads();
  for (int l = tid; l < Lsz; l += 64) {
    int v = s_seq[l] > 0; s_val[l] = (unsigned char)v;
    int iv = 0;
    if (v) { for (int j = 0; j < Lsz; ++j) { if (s_seq[j] > 0 && s_seq[j] == s_seq[l]) { iv = j; break; } } }
    s_inv[l] = iv; inv_g[b * Lsz + l] = iv;
  }
  for (int i = tid; i < Lsz * Lsz; i += 64) { (&sAin[0][0])[i] = 0.f; (&sAout[0][0])[i] = 0.f; }
  __syncthreads();
  if (tid == 0) {
    int len = 0;
    for (int l = 0; l < Lsz; ++l) if (s_val[l]) s_comp[len++] = s_inv[l];
    len_g[b] = len;
    lastn_g[b] = len > 0 ? s_comp[len - 1] : 0;
    for (int i = 0; i + 1 < len; ++i) { sAin[s_comp[i + 1]][s_comp[i]] += 1.f; sAout[s_comp[i]][s_comp[i + 1]] += 1.f; }
  }
  __syncthreads();
  if (tid < Lsz) {
    float a = 0.f, o = 0.f;
    for (int j = 0; j < Lsz; ++j) { a += sAin[tid][j]; o += sAout[tid][j]; }
    rin[tid] = a + 1e-8f; rout[tid] = o + 1e-8f;
  }
  __syncthreads();
  for (int i = tid; i < Lsz * Lsz; i += 64) {
    int r = i / Lsz;
    Ain [(size_t)b * 2500 + i] = (&sAin [0][0])[i] / rin [r];
    Aout[(size_t)b * 2500 + i] = (&sAout[0][0])[i] / rout[r];
  }
  for (int i = tid; i < Lsz * Dsz; i += 64) {
    int l = i >> 7, d = i & 127;
    int used = s_val[l] && (s_inv[l] == l);
    float v = used ? emb[(size_t)s_seq[l] * Dsz + d] : 0.f;
    size_t idx = ((size_t)b * Lsz + l) * Dsz + d;
    h[idx] = v; hbf[idx] = (bf16)v;
  }
}

// m = A_in @ Hin + A_out @ Hout  (per batch, Hin/Hout in LDS)
__global__ void batched_m_kernel(const float* __restrict__ Ain, const float* __restrict__ Aout,
                                 const float* __restrict__ HIO, float* __restrict__ m,
                                 bf16* __restrict__ mbf) {
  const int b = blockIdx.x, tid = threadIdx.x;  // 256 threads
  __shared__ float Hi[Lsz][Dsz], Ho[Lsz][Dsz];
  for (int i = tid; i < Lsz * Dsz; i += 256) {
    int j = i >> 7, d = i & 127;
    size_t r = ((size_t)b * Lsz + j) * 256;
    Hi[j][d] = HIO[r + d]; Ho[j][d] = HIO[r + 128 + d];
  }
  __syncthreads();
  const int d = tid & 127, i0 = tid >> 7;
  for (int i = i0; i < Lsz; i += 2) {
    const float* ai = Ain  + ((size_t)b * Lsz + i) * Lsz;
    const float* ao = Aout + ((size_t)b * Lsz + i) * Lsz;
    float acc = 0.f;
#pragma unroll 5
    for (int j = 0; j < Lsz; ++j) acc += ai[j] * Hi[j][d] + ao[j] * Ho[j][d];
    size_t idx = ((size_t)b * Lsz + i) * Dsz + d;
    m[idx] = acc; mbf[idx] = (bf16)acc;
  }
}

__global__ void gates1_kernel(const float* __restrict__ Gm, const float* __restrict__ Gh,
                              const float* __restrict__ h, const float* __restrict__ b_r,
                              bf16* __restrict__ rhbf) {
  int i = blockIdx.x * 256 + threadIdx.x;
  if (i >= BL * Dsz) return;
  int row = i >> 7, d = i & 127;
  float r = Gm[(size_t)row * 384 + 128 + d] + b_r[d] + Gh[(size_t)row * 256 + 128 + d];
  r = 1.f / (1.f + __expf(-r));
  rhbf[i] = (bf16)(r * h[i]);
}

__global__ void gates2_kernel(const float* __restrict__ Gm, const float* __restrict__ Gh,
                              const float* __restrict__ G6, const float* __restrict__ b_z,
                              const float* __restrict__ b_h, float* __restrict__ h) {
  int i = blockIdx.x * 256 + threadIdx.x;
  if (i >= BL * Dsz) return;
  int row = i >> 7, d = i & 127;
  float z  = Gm[(size_t)row * 384 + d] + b_z[d] + Gh[(size_t)row * 256 + d];
  z = 1.f / (1.f + __expf(-z));
  float ht = tanhf(Gm[(size_t)row * 384 + 256 + d] + b_h[d] + G6[i]);
  h[i] = (1.f - z) * h[i] + z * ht;
}

__global__ void gather_kernel(const float* __restrict__ h, const int* __restrict__ inv,
                              const int* __restrict__ lastn, float* __restrict__ seqh,
                              bf16* __restrict__ seqhbf, float* __restrict__ lasth,
                              bf16* __restrict__ lasthbf) {
  const int b = blockIdx.x, l = blockIdx.y, d = threadIdx.x;
  float v = h[((size_t)b * Lsz + inv[b * Lsz + l]) * Dsz + d];
  size_t idx = ((size_t)b * Lsz + l) * Dsz + d;
  seqh[idx] = v; seqhbf[idx] = (bf16)v;
  if (l == 0) {
    float lv = h[((size_t)b * Lsz + lastn[b]) * Dsz + d];
    lasth[(size_t)b * Dsz + d] = lv; lasthbf[(size_t)b * Dsz + d] = (bf16)lv;
  }
}

__global__ void attn_kernel(const float* __restrict__ Q, const float* __restrict__ K1,
                            const float* __restrict__ b_attk, const float* __restrict__ att_q,
                            const int* __restrict__ seq, const float* __restrict__ seqh,
                            float* __restrict__ sg) {
  const int b = blockIdx.x, d = threadIdx.x;   // 128 threads
  __shared__ float red[Dsz];
  __shared__ float e[Lsz], alpha[Lsz];
  float kd = K1[(size_t)b * Dsz + d] + b_attk[d];
  float qd = att_q[d];
  for (int l = 0; l < Lsz; ++l) {
    red[d] = tanhf(Q[((size_t)b * Lsz + l) * Dsz + d] + kd) * qd;
    __syncthreads();
    for (int s = 64; s > 0; s >>= 1) { if (d < s) red[d] += red[d + s]; __syncthreads(); }
    if (d == 0) e[l] = (seq[b * Lsz + l] > 0) ? red[0] : -1e9f;
    __syncthreads();
  }
  if (d == 0) {
    float mx = -1e30f;
    for (int l = 0; l < Lsz; ++l) mx = fmaxf(mx, e[l]);
    float s = 0.f;
    for (int l = 0; l < Lsz; ++l) { float a = __expf(e[l] - mx); alpha[l] = a; s += a; }
    for (int l = 0; l < Lsz; ++l) alpha[l] /= s;
  }
  __syncthreads();
  float acc = 0.f;
  for (int l = 0; l < Lsz; ++l) acc += alpha[l] * seqh[((size_t)b * Lsz + l) * Dsz + d];
  sg[(size_t)b * Dsz + d] = acc;
}

__global__ void srep_kernel(const float* __restrict__ sg, const float* __restrict__ lasth,
                            const float* __restrict__ Wp, const float* __restrict__ bp,
                            const int* __restrict__ len, bf16* __restrict__ srepbf) {
  const int b = blockIdx.x, d = threadIdx.x;   // 128 threads
  float acc = bp[d];
  for (int k = 0; k < Dsz; ++k) acc += sg[(size_t)b * Dsz + k] * Wp[(size_t)k * Dsz + d];
  for (int k = 0; k < Dsz; ++k) acc += lasth[(size_t)b * Dsz + k] * Wp[(size_t)(Dsz + k) * Dsz + d];
  if (len[b] == 0) acc = 0.f;
  srepbf[(size_t)b * Dsz + d] = (bf16)acc;
}

// ---------------------------------------------------------------------------
extern "C" void kernel_launch(void* const* d_in, const int* in_sizes, int n_in,
                              void* d_out, int out_size, void* d_ws, size_t ws_size,
                              hipStream_t stream) {
  (void)in_sizes; (void)n_in; (void)out_size; (void)ws_size;
  const int*   seq      = (const int*)  d_in[0];
  const float* item_emb = (const float*)d_in[1];
  const float* W_in     = (const float*)d_in[2];
  const float* W_out    = (const float*)d_in[3];
  const float* W_z      = (const float*)d_in[4];
  const float* b_z      = (const float*)d_in[5];
  const float* U_z      = (const float*)d_in[6];
  const float* W_r      = (const float*)d_in[7];
  const float* b_r      = (const float*)d_in[8];
  const float* U_r      = (const float*)d_in[9];
  const float* W_h      = (const float*)d_in[10];
  const float* b_h      = (const float*)d_in[11];
  const float* U_h      = (const float*)d_in[12];
  const float* att_Wq   = (const float*)d_in[13];
  const float* att_Wk   = (const float*)d_in[14];
  const float* b_attk   = (const float*)d_in[15];
  const float* att_q    = (const float*)d_in[16];
  const float* W_proj   = (const float*)d_in[17];
  const float* b_proj   = (const float*)d_in[18];
  float* out = (float*)d_out;
  char*  ws  = (char*)d_ws;

  bf16*  embbf  = (bf16*) (ws + OFF_EMB);
  bf16*  wp     = (bf16*) (ws + OFF_WPACK);
  float* Ain    = (float*)(ws + OFF_AIN);
  float* Aout   = (float*)(ws + OFF_AOUT);
  float* h      = (float*)(ws + OFF_H);
  bf16*  hbf    = (bf16*) (ws + OFF_HBF);
  float* HIO    = (float*)(ws + OFF_HIO);
  float* m      = (float*)(ws + OFF_M);
  bf16*  mbf    = (bf16*) (ws + OFF_MBF);
  float* Gm     = (float*)(ws + OFF_GM);
  float* Gh     = (float*)(ws + OFF_GH);
  bf16*  rhbf   = (bf16*) (ws + OFF_RHBF);
  float* G6     = (float*)(ws + OFF_G6);
  float* seqh   = (float*)(ws + OFF_SEQH);
  bf16*  seqhbf = (bf16*) (ws + OFF_SEQHBF);
  float* Q      = (float*)(ws + OFF_Q);
  float* lasth  = (float*)(ws + OFF_LASTH);
  bf16*  lasthbf= (bf16*) (ws + OFF_LASTHBF);
  float* K1     = (float*)(ws + OFF_K1);
  float* sg     = (float*)(ws + OFF_SG);
  bf16*  srepbf = (bf16*) (ws + OFF_SREPBF);
  int*   invg   = (int*)  (ws + OFF_INV);
  int*   leng   = (int*)  (ws + OFF_LEN);
  int*   lastng = (int*)  (ws + OFF_LASTN);

  // pack weights -> bf16 transposed (Bt[N][K]); order makes [Wz|Wr|Wh], [Uz|Ur] contiguous
  const float* packs[10] = {W_in, W_out, W_z, W_r, W_h, U_z, U_r, U_h, att_Wq, att_Wk};
  for (int i = 0; i < 10; ++i)
    packT_kernel<<<dim3(128), 128, 0, stream>>>(packs[i], wp + (size_t)i * 16384);

  // embedding -> bf16 (L2-resident operand for logits GEMM)
  {
    size_t n = (size_t)Vp1 * Dsz;
    cvt_bf16_kernel<<<dim3((unsigned)((n + 255) / 256)), 256, 0, stream>>>(item_emb, embbf, n);
  }

  preprocess_kernel<<<dim3(Bsz), 64, 0, stream>>>(seq, item_emb, Ain, Aout, h, hbf,
                                                  invg, leng, lastng);

  // [Hin|Hout] = h @ [W_in|W_out]
  wmma_gemm_kernel<<<dim3(BL / 16, 4), 32, 0, stream>>>(hbf, wp, HIO, BL, 256);
  batched_m_kernel<<<dim3(Bsz), 256, 0, stream>>>(Ain, Aout, HIO, m, mbf);
  // Gm = m @ [W_z|W_r|W_h],  Gh = h @ [U_z|U_r]
  wmma_gemm_kernel<<<dim3(BL / 16, 6), 32, 0, stream>>>(mbf, wp + (size_t)2 * 16384, Gm, BL, 384);
  wmma_gemm_kernel<<<dim3(BL / 16, 4), 32, 0, stream>>>(hbf, wp + (size_t)5 * 16384, Gh, BL, 256);
  gates1_kernel<<<dim3((BL * Dsz + 255) / 256), 256, 0, stream>>>(Gm, Gh, h, b_r, rhbf);
  // G6 = (r*h) @ U_h
  wmma_gemm_kernel<<<dim3(BL / 16, 2), 32, 0, stream>>>(rhbf, wp + (size_t)7 * 16384, G6, BL, 128);
  gates2_kernel<<<dim3((BL * Dsz + 255) / 256), 256, 0, stream>>>(Gm, Gh, G6, b_z, b_h, h);

  gather_kernel<<<dim3(Bsz, Lsz), 128, 0, stream>>>(h, invg, lastng, seqh, seqhbf,
                                                    lasth, lasthbf);
  // Q = seq_h @ att_Wq,  K1 = last_h @ att_Wk
  wmma_gemm_kernel<<<dim3(BL / 16, 2), 32, 0, stream>>>(seqhbf, wp + (size_t)8 * 16384, Q, BL, 128);
  wmma_gemm_kernel<<<dim3(Bsz / 16, 2), 32, 0, stream>>>(lasthbf, wp + (size_t)9 * 16384, K1, Bsz, 128);
  attn_kernel<<<dim3(Bsz), 128, 0, stream>>>(Q, K1, b_attk, att_q, seq, seqh, sg);
  srep_kernel<<<dim3(Bsz), 128, 0, stream>>>(sg, lasth, W_proj, b_proj, leng, srepbf);

  // logits = s_rep @ emb^T  (write-bound; 6251 vocab tiles)
  logits_kernel<<<dim3((Vp1 + 15) / 16), 32, 0, stream>>>(srepbf, embbf, out);
}
